// SS2DMambaBlock_31310311588296
// MI455X (gfx1250) — matrix operand order repaired and checked by reading, not compiled, not run on hardware
//
#include <hip/hip_runtime.h>
#include <hip/hip_bf16.h>

// ---------------------------------------------------------------------------
// SS2D Mamba block for MI455X (gfx1250, wave32, WMMA + async global->LDS).
// Shapes: B=8, H=W=64, L=4096, C=256, D_INNER=512, D_STATE=16, DT_RANK=16.
// Tokens M = B*L = 32768.
//   1) LayerNorm -> xn (bf16)
//   2) ONE shared in-proj GEMM (wmma bf16, LDS-staged weights): xz = xn@w_in^T
//      split into xi (f32) and silu(z) (f32)           [direction independent]
//   3) per direction: conv+silu -> xc; dbl = xc@w_xdbl^T (wmma, N=48 direct);
//      dt = softplus(...); selective scan (1 block/seq), gate, accumulate
//   4) mamba = y_sum @ w_out^T (wmma, LDS-staged; single GEMM, w_out shared)
//   5) out = mamba @ proj_w^T + proj_b + residual (wmma, fused epilogue)
// ---------------------------------------------------------------------------

#define TOKENS   32768
#define C_DIM    256
#define DI       512
#define DSTATE   16
#define DTRANK   16
#define SEQ_N    512
#define SEQ_T    64

typedef __attribute__((ext_vector_type(16))) __bf16 bf16x16;
typedef __attribute__((ext_vector_type(8)))  float  f32x8;

__device__ __forceinline__ unsigned short f32_to_bf16(float f) {
    unsigned int u = __float_as_uint(f);
    u += 0x7fffu + ((u >> 16) & 1u);        // round-to-nearest-even
    return (unsigned short)(u >> 16);
}

__device__ __forceinline__ float silu_f(float v) {
    return v * (1.0f / (1.0f + __expf(-v)));
}

// token index in canonical (b,h,w) order for direction-ordered position (seq,t)
__device__ __forceinline__ int tok(int dir, int seq, int t) {
    int b = seq >> 6, r = seq & 63;
    switch (dir) {
        case 0:  return (b << 12) + (r << 6) + t;            // horizontal fwd
        case 1:  return (b << 12) + (r << 6) + (63 - t);     // horizontal bwd
        case 2:  return (b << 12) + (t << 6) + r;            // vertical fwd
        default: return (b << 12) + ((63 - t) << 6) + r;     // vertical bwd
    }
}

// async 16B global->LDS copy (ASYNCcnt); GV addressing: lds-offset VGPR + vaddr64
__device__ __forceinline__ void async_copy_b128(unsigned lds_off, const void* g) {
    unsigned long long ga = (unsigned long long)g;
    asm volatile("global_load_async_to_lds_b128 %0, %1, off"
                 :: "v"(lds_off), "v"(ga) : "memory");
}
__device__ __forceinline__ void wait_async_le1() {
    asm volatile("s_wait_asynccnt 0x1" ::: "memory");
}
__device__ __forceinline__ void wait_async_0() {
    asm volatile("s_wait_asynccnt 0x0" ::: "memory");
}

// --------------------------- utility kernels -------------------------------

__global__ __launch_bounds__(256) void fill_zero_k(float* __restrict__ p, int n) {
    int i = blockIdx.x * 256 + threadIdx.x;
    if (i < n) p[i] = 0.0f;
}

__global__ __launch_bounds__(256) void to_bf16_k(const float* __restrict__ in,
                                                 unsigned short* __restrict__ out, int n) {
    int i = blockIdx.x * 256 + threadIdx.x;
    if (i < n) out[i] = f32_to_bf16(in[i]);
}

// --------------------------- LayerNorm -------------------------------------
__global__ __launch_bounds__(256) void layernorm_k(const float* __restrict__ x,
                                                   const float* __restrict__ g,
                                                   const float* __restrict__ bta,
                                                   unsigned short* __restrict__ out) {
    __shared__ float red[256];
    const int m = blockIdx.x, c = threadIdx.x;
    float v = x[(size_t)m * C_DIM + c];
    red[c] = v; __syncthreads();
    for (int s = 128; s > 0; s >>= 1) { if (c < s) red[c] += red[c + s]; __syncthreads(); }
    float mu = red[0] * (1.0f / C_DIM); __syncthreads();
    float d = v - mu;
    red[c] = d * d; __syncthreads();
    for (int s = 128; s > 0; s >>= 1) { if (c < s) red[c] += red[c + s]; __syncthreads(); }
    float var = red[0] * (1.0f / C_DIM);
    float xn = d * rsqrtf(var + 1e-6f) * g[c] + bta[c];
    out[(size_t)m * C_DIM + c] = f32_to_bf16(xn);
}

// ------------------- epilogue (shared by both GEMM kernels) ----------------
// modes: 0 = store f32
//        1 = split (N=1024): n<512 -> O0 (xi), n>=512 -> O1 = silu(z)
//        2 = store f32 to O0 and bf16 to OB
//        3 = O0 = acc + bias[n] + resid[m*N+n]
__device__ __forceinline__ void epilogue_store(
        int mode, int m, int n, int N, float v,
        float* __restrict__ O0, float* __restrict__ O1,
        unsigned short* __restrict__ OB,
        const float* __restrict__ bias, const float* __restrict__ resid) {
    size_t off = (size_t)m * N + n;
    if (mode == 0) {
        O0[off] = v;
    } else if (mode == 1) {
        if (n < DI) O0[(size_t)m * DI + n] = v;
        else        O1[(size_t)m * DI + (n - DI)] = silu_f(v);
    } else if (mode == 2) {
        O0[off] = v;
        OB[off] = f32_to_bf16(v);
    } else {
        O0[off] = v + bias[n] + resid[off];
    }
}

// ------------------ WMMA GEMM, direct global loads (small N) ---------------
// Y[M,N] = X[M,K] @ W[N,K]^T. One wave per 16x16 tile. Used for N=48.
__global__ __launch_bounds__(256) void gemm_wmma_bf16_k(
        const unsigned short* __restrict__ X, const unsigned short* __restrict__ W,
        int M, int N, int K, int mode,
        float* __restrict__ O0, float* __restrict__ O1,
        unsigned short* __restrict__ OB,
        const float* __restrict__ bias, const float* __restrict__ resid) {
    const int lane = threadIdx.x & 31;
    const int wave = threadIdx.x >> 5;
    const int tilesN = N >> 4;
    const int tilesM = M >> 4;
    int tile = blockIdx.x * 8 + wave;
    if (tile >= tilesM * tilesN) return;
    const int tm = (tile / tilesN) << 4;
    const int tn = (tile % tilesN) << 4;

    const int half = lane >> 4;
    const int l15  = lane & 15;

    const unsigned short* xrow = X + (size_t)(tm + l15) * K;
    const unsigned short* wrow = W + (size_t)(tn + l15) * K;

    union Frag { bf16x16 v; unsigned int u[8]; };
    Frag a, b;
    f32x8 acc = {};

    for (int k0 = 0; k0 < K; k0 += 32) {
        if (k0 + 32 < K) {
            __builtin_prefetch(xrow + k0 + 32, 0, 0);
            __builtin_prefetch(wrow + k0 + 32, 0, 0);
        }
#pragma unroll
        for (int i = 0; i < 8; ++i) {
            // A 16x32 bf16: VGPR i holds K pair at 16*half + 2i
            int ka = k0 + 16 * half + 2 * i;
            a.u[i] = *(const unsigned int*)(xrow + ka);
            // B 32x16 bf16: VGPR i holds K pair at 16*half + 2i
            int kb = k0 + 16 * half + 2 * i;
            b.u[i] = *(const unsigned int*)(wrow + kb);
        }
        acc = __builtin_amdgcn_wmma_f32_16x16x32_bf16(
                  false, a.v, false, b.v, (short)0, acc, false, false);
    }
#pragma unroll
    for (int r = 0; r < 8; ++r)
        epilogue_store(mode, tm + r + 8 * half, tn + l15, N, acc[r],
                       O0, O1, OB, bias, resid);
}

// ---------- WMMA GEMM, weights async-staged in LDS (N % 64 == 0) -----------
// Block = 8 waves -> 128 M-rows x 64 N-cols. Weight strip (64 rows x 32 K of
// bf16 = 4KB) is double-buffered in LDS via global_load_async_to_lds_b128
// (ASYNCcnt / s_wait_asynccnt). Each wave: 1 A-fragment load per K-slab,
// 4 accumulators (16x64), B fragments from LDS (ds_load_b128).
__global__ __launch_bounds__(256) void gemm_wmma_smem_k(
        const unsigned short* __restrict__ X, const unsigned short* __restrict__ W,
        int M, int N, int K, int mode,
        float* __restrict__ O0, float* __restrict__ O1,
        unsigned short* __restrict__ OB,
        const float* __restrict__ bias, const float* __restrict__ resid) {
    __shared__ __align__(16) unsigned short smB[2][64 * 32];   // 2 x 4KB

    const int lane = threadIdx.x & 31;
    const int wave = threadIdx.x >> 5;
    const int half = lane >> 4;
    const int l15  = lane & 15;

    const int nStripsN = N >> 6;
    const int sm = blockIdx.x / nStripsN;       // 128-row M strip
    const int sn = blockIdx.x % nStripsN;       // 64-col  N strip
    const int tm  = sm * 128 + wave * 16;
    const int tn0 = sn * 64;

    // per-thread 16B copy lane: row (0..63) within strip, 16B chunk (0..3)
    const int cpN = threadIdx.x >> 2;
    const int cpC = threadIdx.x & 3;
    const unsigned short* wsrc = W + (size_t)(tn0 + cpN) * K + cpC * 8;
    const unsigned lds0 = (unsigned)(size_t)&smB[0][cpN * 32 + cpC * 8];
    const unsigned lds1 = (unsigned)(size_t)&smB[1][cpN * 32 + cpC * 8];

    const unsigned short* xrow = X + (size_t)(tm + l15) * K;

    union Frag { bf16x16 v; unsigned int u[8]; };
    Frag a, b;
    f32x8 acc0 = {}, acc1 = {}, acc2 = {}, acc3 = {};

    const int nsl = K >> 5;
    async_copy_b128(lds0, wsrc);                       // stage slab 0
    for (int ks = 0; ks < nsl; ++ks) {
        const int buf = ks & 1;
        if (ks + 1 < nsl) {                            // stage next slab
            async_copy_b128(buf ? lds0 : lds1, wsrc + (ks + 1) * 32);
            wait_async_le1();                          // slab ks complete
        } else {
            wait_async_0();
        }
        __syncthreads();                               // LDS visible to all waves

        const int k0 = ks * 32;
#pragma unroll
        for (int i = 0; i < 8; ++i) {                  // A fragment (global)
            int ka = k0 + 16 * half + 2 * i;
            a.u[i] = *(const unsigned int*)(xrow + ka);
        }
#pragma unroll
        for (int j = 0; j < 4; ++j) {                  // 4 B fragments (LDS)
            const int bbase = (16 * j + l15) * 32 + 16 * half;  // 32B contiguous
#pragma unroll
            for (int i = 0; i < 8; ++i)
                b.u[i] = *(const unsigned int*)&smB[buf][bbase + 2 * i];
            if (j == 0) acc0 = __builtin_amdgcn_wmma_f32_16x16x32_bf16(
                                  false, a.v, false, b.v, (short)0, acc0, false, false);
            else if (j == 1) acc1 = __builtin_amdgcn_wmma_f32_16x16x32_bf16(
                                  false, a.v, false, b.v, (short)0, acc1, false, false);
            else if (j == 2) acc2 = __builtin_amdgcn_wmma_f32_16x16x32_bf16(
                                  false, a.v, false, b.v, (short)0, acc2, false, false);
            else             acc3 = __builtin_amdgcn_wmma_f32_16x16x32_bf16(
                                  false, a.v, false, b.v, (short)0, acc3, false, false);
        }
        __syncthreads();                               // protect buf before overwrite
    }

#pragma unroll
    for (int r = 0; r < 8; ++r) {
        const int m = tm + r + 8 * half;
        epilogue_store(mode, m, tn0 +  0 + l15, N, acc0[r], O0, O1, OB, bias, resid);
        epilogue_store(mode, m, tn0 + 16 + l15, N, acc1[r], O0, O1, OB, bias, resid);
        epilogue_store(mode, m, tn0 + 32 + l15, N, acc2[r], O0, O1, OB, bias, resid);
        epilogue_store(mode, m, tn0 + 48 + l15, N, acc3[r], O0, O1, OB, bias, resid);
    }
}

// --------------------- causal depthwise conv + silu ------------------------
__global__ __launch_bounds__(256) void conv_silu_k(
        const float* __restrict__ xi, const float* __restrict__ cw,
        const float* __restrict__ cb, float* __restrict__ xcf,
        unsigned short* __restrict__ xcb, int dir) {
    int idx = blockIdx.x * 256 + threadIdx.x;     // TOKENS*DI threads
    int c = idx & (DI - 1);
    int p = idx >> 9;
    int seq = p >> 6, t = p & 63;
    float acc = cb[c];
#pragma unroll
    for (int k = 0; k < 4; ++k) {
        int tt = t - 3 + k;
        if (tt >= 0)
            acc = fmaf(cw[c * 4 + k], xi[(size_t)tok(dir, seq, tt) * DI + c], acc);
    }
    float s = silu_f(acc);
    xcf[(size_t)p * DI + c] = s;
    xcb[(size_t)p * DI + c] = f32_to_bf16(s);
}

// ------------------------ dt projection + softplus -------------------------
__global__ __launch_bounds__(256) void dt_softplus_k(
        const float* __restrict__ dbl, const float* __restrict__ w_dt,
        const float* __restrict__ b_dt, float* __restrict__ dt) {
    int idx = blockIdx.x * 256 + threadIdx.x;     // TOKENS*DI threads
    int c = idx & (DI - 1);
    int p = idx >> 9;
    float acc = b_dt[c];
#pragma unroll
    for (int r = 0; r < DTRANK; ++r)
        acc = fmaf(dbl[(size_t)p * 48 + r], w_dt[c * DTRANK + r], acc);
    dt[(size_t)p * DI + c] = (acc > 20.0f) ? acc : log1pf(__expf(acc));
}

// ------------------------------ selective scan -----------------------------
__global__ __launch_bounds__(256) void scan_k(
        const float* __restrict__ dbl, const float* __restrict__ dt,
        const float* __restrict__ xc, const float* __restrict__ A_log,
        const float* __restrict__ Dp, const float* __restrict__ sz,
        float* __restrict__ y_sum, int dir) {
    __shared__ float sB[DSTATE], sC[DSTATE];
    const int seq = blockIdx.x;
    const int tid = threadIdx.x;
    const int c0 = tid * 2;

    float A0[DSTATE], A1[DSTATE], h0[DSTATE], h1[DSTATE];
#pragma unroll
    for (int s = 0; s < DSTATE; ++s) {
        A0[s] = -__expf(A_log[c0 * DSTATE + s]);
        A1[s] = -__expf(A_log[(c0 + 1) * DSTATE + s]);
        h0[s] = 0.0f; h1[s] = 0.0f;
    }
    const float dp0 = Dp[c0], dp1 = Dp[c0 + 1];

    for (int t = 0; t < SEQ_T; ++t) {
        const int p = seq * SEQ_T + t;
        if (tid < DSTATE)            sB[tid]          = dbl[(size_t)p * 48 + DTRANK + tid];
        else if (tid < 2 * DSTATE)   sC[tid - DSTATE] = dbl[(size_t)p * 48 + DTRANK + DSTATE + (tid - DSTATE)];
        __syncthreads();

        const float dt0 = dt[(size_t)p * DI + c0];
        const float dt1 = dt[(size_t)p * DI + c0 + 1];
        const float x0  = xc[(size_t)p * DI + c0];
        const float x1  = xc[(size_t)p * DI + c0 + 1];
        const float dtx0 = dt0 * x0, dtx1 = dt1 * x1;
        float y0 = 0.0f, y1 = 0.0f;
#pragma unroll
        for (int s = 0; s < DSTATE; ++s) {
            const float bv = sB[s], cv = sC[s];
            h0[s] = fmaf(__expf(dt0 * A0[s]), h0[s], dtx0 * bv);
            y0 = fmaf(h0[s], cv, y0);
            h1[s] = fmaf(__expf(dt1 * A1[s]), h1[s], dtx1 * bv);
            y1 = fmaf(h1[s], cv, y1);
        }
        const int token = tok(dir, seq, t);
        const size_t o = (size_t)token * DI + c0;
        y_sum[o]     += (y0 + dp0 * x0) * sz[o];      // unique (token,ch) per launch
        y_sum[o + 1] += (y1 + dp1 * x1) * sz[o + 1];
        __syncthreads();
    }
}

// ------------------------------- launcher ----------------------------------

extern "C" void kernel_launch(void* const* d_in, const int* in_sizes, int n_in,
                              void* d_out, int out_size, void* d_ws, size_t ws_size,
                              hipStream_t stream) {
    const float* x      = (const float*)d_in[0];
    const float* norm_g = (const float*)d_in[1];
    const float* norm_b = (const float*)d_in[2];
    const float* w_in   = (const float*)d_in[3];
    const float* conv_w = (const float*)d_in[4];
    const float* conv_b = (const float*)d_in[5];
    const float* w_xdbl = (const float*)d_in[6];
    const float* w_dt   = (const float*)d_in[7];
    const float* b_dt   = (const float*)d_in[8];
    const float* A_log  = (const float*)d_in[9];
    const float* Dp     = (const float*)d_in[10];
    const float* w_out  = (const float*)d_in[11];
    const float* proj_w = (const float*)d_in[12];
    const float* proj_b = (const float*)d_in[13];
    float* out = (float*)d_out;

    char* ws = (char*)d_ws;
    auto alloc = [&](size_t bytes) -> char* {
        char* p = ws; ws += (bytes + 255) & ~(size_t)255; return p;
    };
    unsigned short* xn_b     = (unsigned short*)alloc((size_t)TOKENS * C_DIM * 2);
    unsigned short* w_in_b   = (unsigned short*)alloc((size_t)2 * DI * C_DIM * 2);
    unsigned short* w_xdbl_b = (unsigned short*)alloc((size_t)48 * DI * 2);
    unsigned short* w_out_b  = (unsigned short*)alloc((size_t)C_DIM * DI * 2);
    unsigned short* proj_w_b = (unsigned short*)alloc((size_t)C_DIM * C_DIM * 2);
    float* xi_f    = (float*)alloc((size_t)TOKENS * DI * 4);
    float* sz_f    = (float*)alloc((size_t)TOKENS * DI * 4);       // silu(z)
    float* xc_f    = (float*)alloc((size_t)TOKENS * DI * 4);
    unsigned short* xc_b = (unsigned short*)alloc((size_t)TOKENS * DI * 2);
    float* dbl_f   = (float*)alloc((size_t)TOKENS * 48 * 4);
    float* dt_f    = (float*)alloc((size_t)TOKENS * DI * 4);
    float* y_sum   = (float*)alloc((size_t)TOKENS * DI * 4);
    unsigned short* ysum_b = (unsigned short*)alloc((size_t)TOKENS * DI * 2);
    float* mamba_f = (float*)alloc((size_t)TOKENS * C_DIM * 4);
    unsigned short* mamba_b = (unsigned short*)alloc((size_t)TOKENS * C_DIM * 2);

    // weights -> bf16
    to_bf16_k<<<(2 * DI * C_DIM + 255) / 256, 256, 0, stream>>>(w_in, w_in_b, 2 * DI * C_DIM);
    to_bf16_k<<<(48 * DI + 255) / 256, 256, 0, stream>>>(w_xdbl, w_xdbl_b, 48 * DI);
    to_bf16_k<<<(C_DIM * DI + 255) / 256, 256, 0, stream>>>(w_out, w_out_b, C_DIM * DI);
    to_bf16_k<<<(C_DIM * C_DIM + 255) / 256, 256, 0, stream>>>(proj_w, proj_w_b, C_DIM * C_DIM);

    // LayerNorm
    layernorm_k<<<TOKENS, 256, 0, stream>>>(x, norm_g, norm_b, xn_b);

    // shared in-projection: xz = xn @ w_in^T, split xi / silu(z)  (N=1024)
    gemm_wmma_smem_k<<<(TOKENS / 128) * ((2 * DI) / 64), 256, 0, stream>>>(
        xn_b, w_in_b, TOKENS, 2 * DI, C_DIM, /*mode=*/1,
        xi_f, sz_f, nullptr, nullptr, nullptr);

    // zero the 4-direction accumulator
    fill_zero_k<<<(TOKENS * DI + 255) / 256, 256, 0, stream>>>(y_sum, TOKENS * DI);

    // 4 directional scans
    for (int dir = 0; dir < 4; ++dir) {
        conv_silu_k<<<(TOKENS * DI) / 256, 256, 0, stream>>>(
            xi_f, conv_w, conv_b, xc_f, xc_b, dir);
        int tiles = (TOKENS / 16) * (48 / 16);
        gemm_wmma_bf16_k<<<(tiles + 7) / 8, 256, 0, stream>>>(
            xc_b, w_xdbl_b, TOKENS, 48, DI, /*mode=*/0,
            dbl_f, nullptr, nullptr, nullptr, nullptr);
        dt_softplus_k<<<(TOKENS * DI) / 256, 256, 0, stream>>>(dbl_f, w_dt, b_dt, dt_f);
        scan_k<<<SEQ_N, 256, 0, stream>>>(dbl_f, dt_f, xc_f, A_log, Dp, sz_f, y_sum, dir);
    }

    // out-projection on the pre-summed gated outputs  (N=256)
    to_bf16_k<<<(TOKENS * DI + 255) / 256, 256, 0, stream>>>(y_sum, ysum_b, TOKENS * DI);
    gemm_wmma_smem_k<<<(TOKENS / 128) * (C_DIM / 64), 256, 0, stream>>>(
        ysum_b, w_out_b, TOKENS, C_DIM, DI, /*mode=*/2,
        mamba_f, nullptr, mamba_b, nullptr, nullptr);

    // final projection + bias + residual  (N=256)
    gemm_wmma_smem_k<<<(TOKENS / 128) * (C_DIM / 64), 256, 0, stream>>>(
        mamba_b, proj_w_b, TOKENS, C_DIM, C_DIM, /*mode=*/3,
        out, nullptr, nullptr, proj_b, x);
}
